// GroupedQueryAttention_82386062672301
// MI455X (gfx1250) — matrix operand order repaired and checked by reading, not compiled, not run on hardware
//
#include <hip/hip_runtime.h>

// ---------------------------------------------------------------------------
// GroupedQueryAttention for MI455X (gfx1250, wave32, WMMA bf16 16x16x32)
//
// Pipeline:
//   1) convert x -> bf16; convert+transpose Wq/Wk/Wv/Wo -> bf16 [N][K]
//   2) WMMA GEMM, B tiles staged in LDS double-buffered via TDM
//      (tensor_load_to_lds, TENSORcnt) or async DMA (ASYNCcnt):
//      Q = x@Wq+bq (RoPE fused), K = x@Wk+bk (RoPE fused), V (stored [d][s])
//   3) flash-attention (online softmax, causal), WMMA for QK^T and PV,
//      register-pipelined K/V fragments
//   4) WMMA GEMM: out = attn @ Wo + bo  (fp32 output)
// ---------------------------------------------------------------------------

typedef unsigned short u16;
typedef unsigned long long u64;
typedef u16      v8us  __attribute__((ext_vector_type(8)));
typedef u16      v16us __attribute__((ext_vector_type(16)));
typedef __bf16   v16bf __attribute__((ext_vector_type(16)));
typedef float    v8f   __attribute__((ext_vector_type(8)));
typedef unsigned v4u   __attribute__((ext_vector_type(4)));
typedef unsigned v8u   __attribute__((ext_vector_type(8)));

#define BATCH   2
#define SEQ     1024
#define DMODEL  2048
#define NHEAD   32
#define NKV     8
#define DK      64
#define BSROWS  (BATCH * SEQ)
#define LN_THETA_OVER_HALF 0.28782313805733795f   // ln(10000)/32

// ---- float -> bf16 (round to nearest even) --------------------------------
__device__ __forceinline__ u16 f2bf(float f) {
    unsigned u = __builtin_bit_cast(unsigned, f);
    unsigned r = u + 0x7fffu + ((u >> 16) & 1u);
    return (u16)(r >> 16);
}

__device__ __forceinline__ v16us combine(v8us lo, v8us hi) {
    v16us r;
#pragma unroll
    for (int i = 0; i < 8; ++i) { r[i] = lo[i]; r[8 + i] = hi[i]; }
    return r;
}

// D = A(16x32 bf16) * B(32x16 bf16) + C(16x16 f32)
__device__ __forceinline__ v8f wmma_bf16(v16us a, v16us b, v8f c) {
    return __builtin_amdgcn_wmma_f32_16x16x32_bf16(
        false, __builtin_bit_cast(v16bf, a),
        false, __builtin_bit_cast(v16bf, b),
        (short)0, c, false, false);
}

// A-operand: lane l holds row (l&15); K elems {8h..8h+7, 16+8h..16+8h+7}
__device__ __forceinline__ v16us load_frag_a(const u16* p) {
    return combine(*(const v8us*)p, *(const v8us*)(p + 16));
}
// B-operand: lane l holds col (l&15); K elems {8h..8h+7, 16+8h..16+8h+7}
__device__ __forceinline__ v16us load_frag_b(const u16* p) {
    return combine(*(const v8us*)p, *(const v8us*)(p + 16));
}

// low 32 bits of a generic pointer to __shared__ = wave-relative LDS byte addr
__device__ __forceinline__ unsigned lds_addr_of(const void* p) {
    return (unsigned)(u64)p;
}

// async DMA: 16 bytes global -> LDS, tracked by ASYNCcnt (GVS addressing)
__device__ __forceinline__ void async_copy_b128(unsigned lds_byte, unsigned goff_byte,
                                                u64 gbase) {
    asm volatile("global_load_async_to_lds_b128 %0, %1, %2"
                 :: "v"(lds_byte), "v"(goff_byte), "s"(gbase) : "memory");
}
__device__ __forceinline__ void wait_async0() {
    asm volatile("s_wait_asynccnt 0x0" ::: "memory");
}

// TDM: DMA a 2D tile (tile_dim0=32 elems x tile_dim1=64 rows, bf16) from a
// row-major [N][K] tensor into LDS. D# per CDNA5 ISA.
__device__ __forceinline__ void tdm_load_tile(u64 gaddr, unsigned lds_byte, int K, int N) {
    v4u g0;
    g0[0] = 1u;                                   // count=1, user descriptor
    g0[1] = lds_byte;                             // lds_addr
    g0[2] = (unsigned)gaddr;                      // global_addr[31:0]
    g0[3] = (unsigned)(gaddr >> 32) | (2u << 30); // global_addr[56:32] | type=2
    v8u g1;
    g1[0] = 1u << 16;                             // data_size=1 (2 bytes)
    g1[1] = (unsigned)K << 16;                    // tensor_dim0[15:0]=K
    g1[2] = ((unsigned)N & 0xffffu) << 16;        // dim0 hi=0 | tensor_dim1 lo=N
    g1[3] = 32u << 16;                            // dim1 hi=0 | tile_dim0=32
    g1[4] = 64u;                                  // tile_dim1=64, tile_dim2=0
    g1[5] = (unsigned)K;                          // tensor_dim0_stride lo=K
    g1[6] = 0u;                                   // stride hi / dim1_stride lo
    g1[7] = 0u;
    asm volatile("tensor_load_to_lds %0, %1" :: "s"(g0), "s"(g1) : "memory");
}

// ---------------------------------------------------------------------------
// Conversion kernels
// ---------------------------------------------------------------------------
__global__ void k_f32_to_bf16(const float* __restrict__ in, u16* __restrict__ out, int n) {
    int i = blockIdx.x * blockDim.x + threadIdx.x;
    if (i < n) out[i] = f2bf(in[i]);
}

// in: [K][N] fp32 row-major  ->  out: [N][K] bf16 row-major
__global__ void k_transpose_bf16(const float* __restrict__ in, u16* __restrict__ out,
                                 int K, int N) {
    int i = blockIdx.x * blockDim.x + threadIdx.x;
    if (i < K * N) {
        int k = i / N, n = i - k * N;
        out[(size_t)n * K + k] = f2bf(in[i]);
    }
}

// ---------------------------------------------------------------------------
// WMMA GEMM:  C[M,N] = A[M,K] @ BT[N,K]^T + bias
//   B tile (64 cols x 32 k = 4KB) double-buffered in LDS, filled by either
//   the Tensor Data Mover (USE_TDM=1, one instruction per tile, wave 0 only)
//   or per-lane global_load_async_to_lds_b128 (USE_TDM=0). Compile-time
//   selection keeps the 64-iteration hot loop free of runtime branching.
//   A fragments are register double-buffered across K-steps.
// mode 0: Q (RoPE, bf16, [b][h][s][64])   mode 1: K (RoPE, bf16, [b][kvh][s][64])
// mode 2: V (bf16, transposed [b][kvh][64][s])   mode 3: O (fp32 row-major)
// ---------------------------------------------------------------------------
template <int USE_TDM>
__global__ __launch_bounds__(128) void k_gemm(
    const u16* __restrict__ A, const u16* __restrict__ BT,
    const float* __restrict__ bias,
    u16* __restrict__ outb, float* __restrict__ outf,
    int M, int N, int K, int mode)
{
    __shared__ __align__(16) u16 sB[2][64 * 32];   // [buf][col*32 + k]

    const int tid  = threadIdx.x;
    const int lane = tid & 31;
    const int w    = tid >> 5;
    const int hh   = lane >> 4;
    const int ln   = lane & 15;
    const int rowbase = blockIdx.y * 64 + w * 16;
    const int nbase   = blockIdx.x * 64;

    // per-thread async chunk assignment: 256 x 16B chunks, 2 per thread
    const int c0  = tid * 2, c1 = tid * 2 + 1;
    const int col0 = c0 >> 2, e0 = (c0 & 3) * 8;
    const int col1 = c1 >> 2, e1 = (c1 & 3) * 8;
    const unsigned goff0 = (unsigned)(((nbase + col0) * K + e0) * 2);
    const unsigned goff1 = (unsigned)(((nbase + col1) * K + e1) * 2);
    const unsigned ldsB  = lds_addr_of(&sB[0][0]);
    const unsigned loff0 = (unsigned)((col0 * 32 + e0) * 2);
    const unsigned loff1 = (unsigned)((col1 * 32 + e1) * 2);

    const u64 tile_g0 = (u64)BT + ((u64)nbase * K) * 2;   // tile origin, bytes

    v8f acc[4] = {};
    const u16* arow = A + (size_t)(rowbase + ln) * K;

    // prologue: DMA tile kk=0 into buf 0; preload first A fragment
    if (USE_TDM) {
        if (w == 0) tdm_load_tile(tile_g0, ldsB, K, N);
    } else {
        async_copy_b128(ldsB + loff0, goff0, (u64)BT);
        async_copy_b128(ldsB + loff1, goff1, (u64)BT);
    }
    v16us afrag = load_frag_a(arow + 8 * hh);

    int buf = 0;
    for (int kk = 0; kk < K; kk += 32) {
        if (USE_TDM) __builtin_amdgcn_s_wait_tensorcnt((short)0);
        else         wait_async0();
        __syncthreads();                       // tile(kk) visible; prev buf free
        if (kk + 32 < K) {
            if (USE_TDM) {
                if (w == 0)
                    tdm_load_tile(tile_g0 + (u64)(kk + 32) * 2,
                                  ldsB + (unsigned)((buf ^ 1) * 4096), K, N);
            } else {
                u64 gb = (u64)BT + (u64)(kk + 32) * 2;
                unsigned lo = ldsB + (unsigned)((buf ^ 1) * 4096);
                async_copy_b128(lo + loff0, goff0, gb);
                async_copy_b128(lo + loff1, goff1, gb);
            }
        }
        // register-pipeline next A fragment (wait lands after the WMMAs)
        v16us anext = afrag;
        if (kk + 32 < K) anext = load_frag_a(arow + kk + 32 + 8 * hh);

        const u16* sb = &sB[buf][8 * hh];
#pragma unroll
        for (int j = 0; j < 4; ++j)
            acc[j] = wmma_bf16(afrag, load_frag_b(sb + (j * 16 + ln) * 32), acc[j]);
        afrag = anext;
        buf ^= 1;
    }

#pragma unroll
    for (int j = 0; j < 4; ++j) {
#pragma unroll
        for (int e = 0; e < 8; ++e) {
            int row = rowbase + e + 8 * hh;          // C layout: M = e + 8*half
            int col = nbase + j * 16 + ln;           //           N = lane&15
            float val = acc[j][e] + bias[col];
            if (mode == 3) {
                outf[(size_t)row * N + col] = val;
            } else if (mode == 2) {
                int b = row >> 10, s = row & (SEQ - 1);
                int kvh = col >> 6, dd = col & 63;
                outb[(((size_t)(b * NKV + kvh) * DK + dd) << 10) + s] = f2bf(val);
            } else {
                // fused RoPE: even/odd head-dim pair lives in lane^1
                int b = row >> 10, s = row & (SEQ - 1);
                int dd = col & 63, jj = dd >> 1;
                float ang = (float)s * __expf(-(float)jj * LN_THETA_OVER_HALF);
                float cs = __cosf(ang), sn = __sinf(ang);
                float other = __shfl_xor(val, 1, 32);
                float r = (lane & 1) ? (other * sn + val * cs)
                                     : (val * cs - other * sn);
                if (mode == 0) {
                    int hq = col >> 6;
                    outb[((((size_t)(b * NHEAD + hq) << 10) + s) << 6) + dd] = f2bf(r);
                } else {
                    int kvh = col >> 6;
                    outb[((((size_t)(b * NKV + kvh) << 10) + s) << 6) + dd] = f2bf(r);
                }
            }
        }
    }
}

// ---------------------------------------------------------------------------
// Flash attention: one wave per 16-query tile; online softmax over 32-key
// blocks; K fragments register-pipelined one block ahead, V fragments issued
// before the softmax section so the shuffle/exp work hides their latency.
// Qb: [b*32+h][s][64]  Kb: [b*8+kvh][s][64]  Vt: [b*8+kvh][64][s]
// Ob: [b][s][2048] bf16 (heads concatenated)
// ---------------------------------------------------------------------------
__global__ __launch_bounds__(128) void k_attn(
    const u16* __restrict__ Qb, const u16* __restrict__ Kb,
    const u16* __restrict__ Vt, u16* __restrict__ Ob)
{
    __shared__ __align__(16) u16 sP[4][16 * 32];    // per-wave P-tile staging
    const int lane = threadIdx.x & 31;
    const int w    = threadIdx.x >> 5;
    const int hh   = lane >> 4, ln = lane & 15;

    const int tile = blockIdx.x * 4 + w;            // 0 .. 4095
    const int bh   = tile >> 6;                     // b*32 + h
    const int q0   = (tile & 63) << 4;
    const int b    = bh >> 5, h = bh & 31;
    const int bkv  = b * NKV + (h >> 2);            // N_REP = 4

    const u16* qrow = Qb + ((((size_t)bh << 10) + q0 + ln) << 6);
    const v16us aq0 = load_frag_a(qrow + 8 * hh);
    const v16us aq1 = load_frag_a(qrow + 32 + 8 * hh);

    const u16* kbase = Kb + ((size_t)bkv << 16);
    const u16* vbase = Vt + ((size_t)bkv << 16);

    float m[8], lsum[8];
    v8f o[4] = {};
#pragma unroll
    for (int e = 0; e < 8; ++e) { m[e] = -1e30f; lsum[e] = 0.f; }

    // preload K fragments for the first 32-key block (two 16-key halves)
    v16us kf00, kf01, kf10, kf11;
    {
        const u16* pk = kbase + (((size_t)ln) << 6) + 8 * hh;
        kf00 = load_frag_b(pk);      kf01 = load_frag_b(pk + 32);
        const u16* pk2 = kbase + (((size_t)(16 + ln)) << 6) + 8 * hh;
        kf10 = load_frag_b(pk2);     kf11 = load_frag_b(pk2 + 32);
    }

    const int kend = q0 + 16;
    for (int kb = 0; kb < kend; kb += 32) {
        v8f cc0 = {}, cc1 = {};
        cc0 = wmma_bf16(aq0, kf00, cc0);
        cc0 = wmma_bf16(aq1, kf01, cc0);
        cc1 = wmma_bf16(aq0, kf10, cc1);
        cc1 = wmma_bf16(aq1, kf11, cc1);

        // issue V fragment loads now; softmax below hides their latency
        v16us vf[4];
#pragma unroll
        for (int j = 0; j < 4; ++j)
            vf[j] = load_frag_b(vbase + (((size_t)(j * 16 + ln)) << 10) + kb + 8 * hh);

        // prefetch next block's K fragments (consumed next iteration)
        if (kb + 32 < kend) {
            const u16* pk = kbase + (((size_t)(kb + 32 + ln)) << 6) + 8 * hh;
            kf00 = load_frag_b(pk);  kf01 = load_frag_b(pk + 32);
            const u16* pk2 = kbase + (((size_t)(kb + 48 + ln)) << 6) + 8 * hh;
            kf10 = load_frag_b(pk2); kf11 = load_frag_b(pk2 + 32);
        }

        float p0[8], p1[8];
#pragma unroll
        for (int e = 0; e < 8; ++e) {
            int row = q0 + e + 8 * hh;
            float s0 = cc0[e] * 0.125f;             // 1/sqrt(64)
            float s1 = cc1[e] * 0.125f;
            if (kb + ln > row)       s0 = -1e30f;
            if (kb + 16 + ln > row)  s1 = -1e30f;
            float mm = fmaxf(s0, s1);
#pragma unroll
            for (int off = 1; off < 16; off <<= 1)
                mm = fmaxf(mm, __shfl_xor(mm, off, 32));
            float mnew = fmaxf(m[e], mm);
            float sc = __expf(m[e] - mnew);
            float e0x = __expf(s0 - mnew);
            float e1x = __expf(s1 - mnew);
            float rs = e0x + e1x;
#pragma unroll
            for (int off = 1; off < 16; off <<= 1)
                rs += __shfl_xor(rs, off, 32);
            lsum[e] = lsum[e] * sc + rs;
            m[e] = mnew;
#pragma unroll
            for (int j = 0; j < 4; ++j) o[j][e] *= sc;
            p0[e] = e0x; p1[e] = e1x;
        }

        // C-fragment -> A-fragment relayout of P through LDS (wave-private)
#pragma unroll
        for (int e = 0; e < 8; ++e) {
            sP[w][(e + 8 * hh) * 32 + ln]      = f2bf(p0[e]);
            sP[w][(e + 8 * hh) * 32 + 16 + ln] = f2bf(p1[e]);
        }
        asm volatile("s_wait_dscnt 0x0" ::: "memory");
        const u16* sprow = &sP[w][ln * 32 + 8 * hh];
        v16us pa = combine(*(const v8us*)sprow, *(const v8us*)(sprow + 16));

#pragma unroll
        for (int j = 0; j < 4; ++j)
            o[j] = wmma_bf16(pa, vf[j], o[j]);
    }

#pragma unroll
    for (int j = 0; j < 4; ++j) {
#pragma unroll
        for (int e = 0; e < 8; ++e) {
            int s = q0 + e + 8 * hh;
            int col = h * DK + j * 16 + ln;
            float val = o[j][e] / lsum[e];
            Ob[(((size_t)b << 10) + s) * DMODEL + col] = f2bf(val);
        }
    }
}

// ---------------------------------------------------------------------------
// Host launch
// ---------------------------------------------------------------------------
extern "C" void kernel_launch(void* const* d_in, const int* in_sizes, int n_in,
                              void* d_out, int out_size, void* d_ws, size_t ws_size,
                              hipStream_t stream) {
    const float* x  = (const float*)d_in[0];
    const float* Wq = (const float*)d_in[1];
    const float* bq = (const float*)d_in[2];
    const float* Wk = (const float*)d_in[3];
    const float* bk = (const float*)d_in[4];
    const float* Wv = (const float*)d_in[5];
    const float* bv = (const float*)d_in[6];
    const float* Wo = (const float*)d_in[7];
    const float* bo = (const float*)d_in[8];
    float* out = (float*)d_out;

    // workspace layout (bytes, 256-aligned). Total ~48 MB.
    char* ws = (char*)d_ws;
    u16* xb  = (u16*)(ws + 0);           //  8 MB  x bf16        [2048][2048]
    u16* WqT = (u16*)(ws + 8388608);     //  8 MB  Wq^T bf16     [2048][2048]
    u16* WkT = (u16*)(ws + 16777216);    //  2 MB  Wk^T bf16     [512][2048]
    u16* WvT = (u16*)(ws + 18874368);    //  2 MB  Wv^T bf16     [512][2048]
    u16* WoT = (u16*)(ws + 20971520);    //  8 MB  Wo^T bf16     [2048][2048]
    u16* Qb  = (u16*)(ws + 29360128);    //  8 MB  roped Q bf16  [b*32+h][s][64]
    u16* Kb  = (u16*)(ws + 37748736);    //  2 MB  roped K bf16  [b*8+kv][s][64]
    u16* Vt  = (u16*)(ws + 39845888);    //  2 MB  V^T bf16      [b*8+kv][64][s]
    u16* Ob  = (u16*)(ws + 41943040);    //  8 MB  attn out bf16 [b][s][2048]

    const int TPB = 256;
    k_f32_to_bf16 <<<(BSROWS * DMODEL + TPB - 1) / TPB, TPB, 0, stream>>>(x, xb, BSROWS * DMODEL);
    k_transpose_bf16<<<(DMODEL * DMODEL + TPB - 1) / TPB, TPB, 0, stream>>>(Wq, WqT, DMODEL, DMODEL);
    k_transpose_bf16<<<(DMODEL * 512   + TPB - 1) / TPB, TPB, 0, stream>>>(Wk, WkT, DMODEL, 512);
    k_transpose_bf16<<<(DMODEL * 512   + TPB - 1) / TPB, TPB, 0, stream>>>(Wv, WvT, DMODEL, 512);
    k_transpose_bf16<<<(DMODEL * DMODEL + TPB - 1) / TPB, TPB, 0, stream>>>(Wo, WoT, DMODEL, DMODEL);

    // QKV projections: TDM-fed B tiles (clean s_wait_tensorcnt loop)
    k_gemm<1><<<dim3(DMODEL / 64, BSROWS / 64), 128, 0, stream>>>(xb, WqT, bq, Qb, nullptr,
                                                                  BSROWS, DMODEL, DMODEL, 0);
    k_gemm<1><<<dim3(512 / 64, BSROWS / 64), 128, 0, stream>>>(xb, WkT, bk, Kb, nullptr,
                                                               BSROWS, 512, DMODEL, 1);
    k_gemm<1><<<dim3(512 / 64, BSROWS / 64), 128, 0, stream>>>(xb, WvT, bv, Vt, nullptr,
                                                               BSROWS, 512, DMODEL, 2);

    k_attn<<<1024, 128, 0, stream>>>(Qb, Kb, Vt, Ob);

    // output projection: async-DMA-fed B tiles (s_wait_asynccnt loop)
    k_gemm<0><<<dim3(DMODEL / 64, BSROWS / 64), 128, 0, stream>>>(Ob, WoT, bo, nullptr, out,
                                                                  BSROWS, DMODEL, DMODEL, 3);
}
